// Attention_49134425866421
// MI455X (gfx1250) — compile-verified, matrix operand order checked by reading, not run on hardware
//
#include <hip/hip_runtime.h>
#include <hip/hip_bf16.h>
#include <math.h>

#define Hh     12
#define SEQ    2048
#define CDIM   768
#define NQKV   2304
#define DHEAD  64
#define NSPEC  5
#define BATCH  4

typedef __bf16 bf16_t;
typedef __attribute__((ext_vector_type(16))) __bf16 v16bf;
typedef __attribute__((ext_vector_type(8)))  __bf16 v8bf;
typedef __attribute__((ext_vector_type(8)))  float  v8f;

// Load 8 contiguous bf16 (16 bytes) into half of a v16bf A/B fragment.
__device__ __forceinline__ void load8(v16bf& dst, int off, const bf16_t* p) {
  v8bf t = *(const v8bf*)p;
#pragma unroll
  for (int i = 0; i < 8; ++i) dst[off + i] = t[i];
}

// ---------------------------------------------------------------------------
// Kernel 1: qkv = x @ W_qkv + b_qkv, RoPE on q/k, scatter to q/k (row-major
// [bh][t][d], bf16) and v transposed ([bh][d][t], bf16).
// Block 256 thr (8 waves), tile 128x128, K-step 32 (one bf16 WMMA K).
// ---------------------------------------------------------------------------
__global__ __launch_bounds__(256) void qkv_rope_kernel(
    const float* __restrict__ x, const float* __restrict__ Wqkv,
    const float* __restrict__ bqkv, const float* __restrict__ rcos,
    const float* __restrict__ rsin, bf16_t* __restrict__ qb,
    bf16_t* __restrict__ kbuf, bf16_t* __restrict__ vtb) {
  __shared__ alignas(16) bf16_t As[128 * 40];
  __shared__ alignas(16) bf16_t Bs[128 * 40];

  const int tid  = threadIdx.x;
  const int lane = tid & 31;
  const int wid  = tid >> 5;
  const int wm   = wid & 3;       // wave M position (4)
  const int wn   = wid >> 2;      // wave N position (2)
  const int m0   = blockIdx.x * 128;
  const int n0   = blockIdx.y * 128;
  const int l16  = lane & 15;
  const int kb0  = (lane < 16) ? 0 : 8;  // 16-bit A/B fragment K-group base
  const int hi8  = (lane >> 4) * 8;      // C fragment row offset per half

  v8f c[2][4];
#pragma unroll
  for (int mi = 0; mi < 2; ++mi)
#pragma unroll
    for (int ni = 0; ni < 4; ++ni)
#pragma unroll
      for (int j = 0; j < 8; ++j) c[mi][ni][j] = 0.0f;

  for (int k0 = 0; k0 < CDIM; k0 += 32) {
    __syncthreads();
    // stage A tile (128x32 f32 -> bf16), 4 float4 per thread
#pragma unroll
    for (int i = 0; i < 4; ++i) {
      int lin = tid + i * 256;               // 0..1023
      int r = lin >> 3, c4 = (lin & 7) * 4;
      float4 f = *(const float4*)(x + (size_t)(m0 + r) * CDIM + k0 + c4);
      bf16_t* d = &As[r * 40 + c4];
      d[0] = (bf16_t)f.x; d[1] = (bf16_t)f.y; d[2] = (bf16_t)f.z; d[3] = (bf16_t)f.w;
    }
    // stage B tile transposed ([n][k]) 32x128 f32 -> bf16
#pragma unroll
    for (int i = 0; i < 4; ++i) {
      int lin = tid + i * 256;
      int kr = lin >> 5, nc = (lin & 31) * 4;
      float4 f = *(const float4*)(Wqkv + (size_t)(k0 + kr) * NQKV + n0 + nc);
      Bs[(nc + 0) * 40 + kr] = (bf16_t)f.x;
      Bs[(nc + 1) * 40 + kr] = (bf16_t)f.y;
      Bs[(nc + 2) * 40 + kr] = (bf16_t)f.z;
      Bs[(nc + 3) * 40 + kr] = (bf16_t)f.w;
    }
    __syncthreads();

    v16bf a[2], b[4];
#pragma unroll
    for (int mi = 0; mi < 2; ++mi) {
      const bf16_t* p = &As[(wm * 32 + mi * 16 + l16) * 40 + kb0];
      load8(a[mi], 0, p); load8(a[mi], 8, p + 16);
    }
#pragma unroll
    for (int ni = 0; ni < 4; ++ni) {
      const bf16_t* p = &Bs[(wn * 64 + ni * 16 + l16) * 40 + kb0];
      load8(b[ni], 0, p); load8(b[ni], 8, p + 16);
    }
#pragma unroll
    for (int mi = 0; mi < 2; ++mi)
#pragma unroll
      for (int ni = 0; ni < 4; ++ni)
        c[mi][ni] = __builtin_amdgcn_wmma_f32_16x16x32_bf16(
            false, a[mi], false, b[ni], (short)0, c[mi][ni], false, false);
  }

  // Epilogue: bias, RoPE (q/k only), scatter.
  const int ncw = n0 + wn * 64;            // 64-aligned -> one head, one of q/k/v
  const int qi  = ncw / CDIM;              // 0=q 1=k 2=v
  const int h   = (ncw % CDIM) / DHEAD;

#pragma unroll
  for (int mi = 0; mi < 2; ++mi) {
    float vals[4][8];
#pragma unroll
    for (int ni = 0; ni < 4; ++ni) {
      float bias = bqkv[ncw + ni * 16 + l16];
#pragma unroll
      for (int j = 0; j < 8; ++j) vals[ni][j] = c[mi][ni][j] + bias;
    }
    int rowbase = m0 + wm * 32 + mi * 16 + hi8;
    if (qi < 2) {
      float outv[4][8];
#pragma unroll
      for (int ni = 0; ni < 4; ++ni) {
        int d = ni * 16 + l16;
#pragma unroll
        for (int j = 0; j < 8; ++j) {
          int t = (rowbase + j) & (SEQ - 1);
          float v = vals[ni][j];
          if (t >= NSPEC) {
            float cs = rcos[(size_t)(t - NSPEC) * DHEAD + d];
            float sn = rsin[(size_t)(t - NSPEC) * DHEAD + d];
            float partner = (ni < 2) ? -vals[ni + 2][j] : vals[ni - 2][j];
            v = v * cs + partner * sn;
          }
          outv[ni][j] = v;
        }
      }
      bf16_t* dst = (qi == 0) ? qb : kbuf;
#pragma unroll
      for (int ni = 0; ni < 4; ++ni) {
        int d = ni * 16 + l16;
#pragma unroll
        for (int j = 0; j < 8; ++j) {
          int row = rowbase + j;
          int bidx = row >> 11, t = row & (SEQ - 1);
          dst[(((size_t)bidx * Hh + h) * SEQ + t) * DHEAD + d] = (bf16_t)outv[ni][j];
        }
      }
    } else {
#pragma unroll
      for (int ni = 0; ni < 4; ++ni) {
        int d = ni * 16 + l16;
#pragma unroll
        for (int j = 0; j < 8; ++j) {
          int row = rowbase + j;
          int bidx = row >> 11, t = row & (SEQ - 1);
          vtb[(((size_t)bidx * Hh + h) * DHEAD + d) * SEQ + t] = (bf16_t)vals[ni][j];
        }
      }
    }
  }
}

// ---------------------------------------------------------------------------
// Kernel 2: flash attention per (b,h). Block 128 thr (4 waves), 128 q-rows
// per block (32 per wave), K-tile 64. Online softmax; P re-fragmented via LDS.
// ---------------------------------------------------------------------------
__global__ __launch_bounds__(128) void attn_kernel(
    const bf16_t* __restrict__ qb, const bf16_t* __restrict__ kbuf,
    const bf16_t* __restrict__ vtb, bf16_t* __restrict__ ao) {
  __shared__ alignas(16) bf16_t Ps[4][32 * 72];

  const int tid  = threadIdx.x;
  const int lane = tid & 31;
  const int wid  = tid >> 5;
  const int bh   = blockIdx.x;            // 0..47
  const int mt   = blockIdx.y;            // 0..15
  const int b    = bh / Hh, h = bh % Hh;
  const int l16  = lane & 15;
  const int kb0  = (lane < 16) ? 0 : 8;
  const int hi8  = (lane >> 4) * 8;
  const int q0   = mt * 128 + wid * 32;

  const bf16_t* qptr = qb   + (size_t)bh * SEQ * DHEAD;
  const bf16_t* kptr = kbuf + (size_t)bh * SEQ * DHEAD;
  const bf16_t* vptr = vtb  + (size_t)bh * DHEAD * SEQ;
  bf16_t* psw = &Ps[wid][0];

  // Q A-fragments resident for whole kernel
  v16bf qa[2][2];
#pragma unroll
  for (int mi = 0; mi < 2; ++mi)
#pragma unroll
    for (int kf = 0; kf < 2; ++kf) {
      const bf16_t* p = qptr + (size_t)(q0 + mi * 16 + l16) * DHEAD + kf * 32 + kb0;
      load8(qa[mi][kf], 0, p); load8(qa[mi][kf], 8, p + 16);
    }

  v8f o[2][4];
  float mrun[2][8], lrun[2][8];
#pragma unroll
  for (int mi = 0; mi < 2; ++mi) {
#pragma unroll
    for (int nd = 0; nd < 4; ++nd)
#pragma unroll
      for (int j = 0; j < 8; ++j) o[mi][nd][j] = 0.0f;
#pragma unroll
    for (int j = 0; j < 8; ++j) { mrun[mi][j] = -3.0e38f; lrun[mi][j] = 0.0f; }
  }

  const float scale = 0.125f;  // 1/sqrt(64)

  for (int kt = 0; kt < SEQ; kt += 64) {
    // S = scale * Q K^T   (K B-fragments straight from global: contiguous)
    v8f s[2][4];
#pragma unroll
    for (int mi = 0; mi < 2; ++mi)
#pragma unroll
      for (int ni = 0; ni < 4; ++ni)
#pragma unroll
        for (int j = 0; j < 8; ++j) s[mi][ni][j] = 0.0f;

#pragma unroll
    for (int kf = 0; kf < 2; ++kf) {
      v16bf kf4[4];
#pragma unroll
      for (int ni = 0; ni < 4; ++ni) {
        const bf16_t* p = kptr + (size_t)(kt + ni * 16 + l16) * DHEAD + kf * 32 + kb0;
        load8(kf4[ni], 0, p); load8(kf4[ni], 8, p + 16);
      }
#pragma unroll
      for (int mi = 0; mi < 2; ++mi)
#pragma unroll
        for (int ni = 0; ni < 4; ++ni)
          s[mi][ni] = __builtin_amdgcn_wmma_f32_16x16x32_bf16(
              false, qa[mi][kf], false, kf4[ni], (short)0, s[mi][ni], false, false);
    }
#pragma unroll
    for (int mi = 0; mi < 2; ++mi)
#pragma unroll
      for (int ni = 0; ni < 4; ++ni)
#pragma unroll
        for (int j = 0; j < 8; ++j) s[mi][ni][j] *= scale;

    // online softmax update (row spans 16 lanes of a half-wave)
#pragma unroll
    for (int mi = 0; mi < 2; ++mi)
#pragma unroll
      for (int j = 0; j < 8; ++j) {
        float rm = fmaxf(fmaxf(s[mi][0][j], s[mi][1][j]),
                         fmaxf(s[mi][2][j], s[mi][3][j]));
#pragma unroll
        for (int off = 1; off < 16; off <<= 1) rm = fmaxf(rm, __shfl_xor(rm, off, 32));
        float mnew = fmaxf(mrun[mi][j], rm);
        float corr = __expf(mrun[mi][j] - mnew);
        mrun[mi][j] = mnew;
        float psum = 0.0f;
#pragma unroll
        for (int ni = 0; ni < 4; ++ni) {
          float pv = __expf(s[mi][ni][j] - mnew);
          s[mi][ni][j] = pv;
          psum += pv;
        }
#pragma unroll
        for (int off = 1; off < 16; off <<= 1) psum += __shfl_xor(psum, off, 32);
        lrun[mi][j] = lrun[mi][j] * corr + psum;
#pragma unroll
        for (int nd = 0; nd < 4; ++nd) o[mi][nd][j] *= corr;
      }

    // C-layout P -> LDS (bf16) -> A-layout fragments
#pragma unroll
    for (int mi = 0; mi < 2; ++mi)
#pragma unroll
      for (int ni = 0; ni < 4; ++ni)
#pragma unroll
        for (int j = 0; j < 8; ++j)
          psw[(mi * 16 + hi8 + j) * 72 + ni * 16 + l16] = (bf16_t)s[mi][ni][j];
    __syncthreads();

    v16bf pa[2][2];
#pragma unroll
    for (int mi = 0; mi < 2; ++mi)
#pragma unroll
      for (int kf = 0; kf < 2; ++kf) {
        const bf16_t* p = &psw[(mi * 16 + l16) * 72 + kf * 32 + kb0];
        load8(pa[mi][kf], 0, p); load8(pa[mi][kf], 8, p + 16);
      }

    // O += P V   (V^T rows -> contiguous B-fragment loads)
#pragma unroll
    for (int kf = 0; kf < 2; ++kf) {
      v16bf vf[4];
#pragma unroll
      for (int nd = 0; nd < 4; ++nd) {
        const bf16_t* p = vptr + (size_t)(nd * 16 + l16) * SEQ + kt + kf * 32 + kb0;
        load8(vf[nd], 0, p); load8(vf[nd], 8, p + 16);
      }
#pragma unroll
      for (int mi = 0; mi < 2; ++mi)
#pragma unroll
        for (int nd = 0; nd < 4; ++nd)
          o[mi][nd] = __builtin_amdgcn_wmma_f32_16x16x32_bf16(
              false, pa[mi][kf], false, vf[nd], (short)0, o[mi][nd], false, false);
    }
    __syncthreads();
  }

  // normalize and write bf16 [b][t][h*64+d]
#pragma unroll
  for (int mi = 0; mi < 2; ++mi)
#pragma unroll
    for (int j = 0; j < 8; ++j) {
      float inv = 1.0f / lrun[mi][j];
      int t = q0 + mi * 16 + hi8 + j;
#pragma unroll
      for (int nd = 0; nd < 4; ++nd) {
        int d = nd * 16 + l16;
        ao[((size_t)b * SEQ + t) * CDIM + h * DHEAD + d] = (bf16_t)(o[mi][nd][j] * inv);
      }
    }
}

// ---------------------------------------------------------------------------
// Kernel 3: out = ao(bf16) @ W_proj + b_proj  (fp32 out). Same tiling as K1.
// ---------------------------------------------------------------------------
__global__ __launch_bounds__(256) void proj_kernel(
    const bf16_t* __restrict__ ao, const float* __restrict__ Wp,
    const float* __restrict__ bp, float* __restrict__ out) {
  __shared__ alignas(16) bf16_t As[128 * 40];
  __shared__ alignas(16) bf16_t Bs[128 * 40];

  const int tid  = threadIdx.x;
  const int lane = tid & 31;
  const int wid  = tid >> 5;
  const int wm   = wid & 3;
  const int wn   = wid >> 2;
  const int m0   = blockIdx.x * 128;
  const int n0   = blockIdx.y * 128;
  const int l16  = lane & 15;
  const int kb0  = (lane < 16) ? 0 : 8;
  const int hi8  = (lane >> 4) * 8;

  v8f c[2][4];
#pragma unroll
  for (int mi = 0; mi < 2; ++mi)
#pragma unroll
    for (int ni = 0; ni < 4; ++ni)
#pragma unroll
      for (int j = 0; j < 8; ++j) c[mi][ni][j] = 0.0f;

  for (int k0 = 0; k0 < CDIM; k0 += 32) {
    __syncthreads();
#pragma unroll
    for (int i = 0; i < 2; ++i) {        // A tile: 128x32 bf16, 16B chunks
      int lin = tid + i * 256;           // 0..511
      int r = lin >> 2, cc = (lin & 3) * 8;
      v8bf t = *(const v8bf*)(ao + (size_t)(m0 + r) * CDIM + k0 + cc);
      *(v8bf*)&As[r * 40 + cc] = t;
    }
#pragma unroll
    for (int i = 0; i < 4; ++i) {        // B tile transposed, f32 -> bf16
      int lin = tid + i * 256;
      int kr = lin >> 5, nc = (lin & 31) * 4;
      float4 f = *(const float4*)(Wp + (size_t)(k0 + kr) * CDIM + n0 + nc);
      Bs[(nc + 0) * 40 + kr] = (bf16_t)f.x;
      Bs[(nc + 1) * 40 + kr] = (bf16_t)f.y;
      Bs[(nc + 2) * 40 + kr] = (bf16_t)f.z;
      Bs[(nc + 3) * 40 + kr] = (bf16_t)f.w;
    }
    __syncthreads();

    v16bf a[2], b[4];
#pragma unroll
    for (int mi = 0; mi < 2; ++mi) {
      const bf16_t* p = &As[(wm * 32 + mi * 16 + l16) * 40 + kb0];
      load8(a[mi], 0, p); load8(a[mi], 8, p + 16);
    }
#pragma unroll
    for (int ni = 0; ni < 4; ++ni) {
      const bf16_t* p = &Bs[(wn * 64 + ni * 16 + l16) * 40 + kb0];
      load8(b[ni], 0, p); load8(b[ni], 8, p + 16);
    }
#pragma unroll
    for (int mi = 0; mi < 2; ++mi)
#pragma unroll
      for (int ni = 0; ni < 4; ++ni)
        c[mi][ni] = __builtin_amdgcn_wmma_f32_16x16x32_bf16(
            false, a[mi], false, b[ni], (short)0, c[mi][ni], false, false);
  }

#pragma unroll
  for (int mi = 0; mi < 2; ++mi)
#pragma unroll
    for (int ni = 0; ni < 4; ++ni) {
      int gn = n0 + wn * 64 + ni * 16 + l16;
      float bias = bp[gn];
#pragma unroll
      for (int j = 0; j < 8; ++j) {
        int row = m0 + wm * 32 + mi * 16 + hi8 + j;
        out[(size_t)row * CDIM + gn] = c[mi][ni][j] + bias;
      }
    }
}

// ---------------------------------------------------------------------------
extern "C" void kernel_launch(void* const* d_in, const int* in_sizes, int n_in,
                              void* d_out, int out_size, void* d_ws, size_t ws_size,
                              hipStream_t stream) {
  const float* x    = (const float*)d_in[0];
  const float* rcos = (const float*)d_in[1];
  const float* rsin = (const float*)d_in[2];
  const float* Wqkv = (const float*)d_in[3];
  const float* bqkv = (const float*)d_in[4];
  const float* Wp   = (const float*)d_in[5];
  const float* bp   = (const float*)d_in[6];
  float* out = (float*)d_out;

  const size_t per = (size_t)BATCH * Hh * SEQ * DHEAD;  // 6.29M elems
  bf16_t* qb  = (bf16_t*)d_ws;
  bf16_t* kb  = qb + per;
  bf16_t* vtb = kb + per;
  bf16_t* ao  = vtb + per;   // BATCH*SEQ*CDIM == per as well

  dim3 g1(8192 / 128, NQKV / 128);   // 64 x 18
  qkv_rope_kernel<<<g1, dim3(256), 0, stream>>>(x, Wqkv, bqkv, rcos, rsin, qb, kb, vtb);

  dim3 g2(BATCH * Hh, SEQ / 128);    // 48 x 16
  attn_kernel<<<g2, dim3(128), 0, stream>>>(qb, kb, vtb, ao);

  dim3 g3(8192 / 128, CDIM / 128);   // 64 x 6
  proj_kernel<<<g3, dim3(256), 0, stream>>>(ao, Wp, bp, out);
}
